// XDecoder_21801253994693
// MI455X (gfx1250) — matrix operand order repaired and checked by reading, not compiled, not run on hardware
//
#include <hip/hip_runtime.h>
#include <hip/hip_bf16.h>
#include <math.h>

// ---------------------------------------------------------------------------
// Model constants (match reference)
// ---------------------------------------------------------------------------
#define B_    64
#define T_    256
#define CTX_  256
#define DIM_  512
#define HEADS_ 8
#define DH_   64
#define DEPTH_ 6
#define EMB_DIM_ 500
#define NCLUS_ 4
#define WSZ_  64
#define MT_   (B_ * T_)        // 16384 rows for every dense GEMM
#define COMMIT_ 0.0001f

typedef __attribute__((ext_vector_type(16))) __bf16 v16bf;
typedef __attribute__((ext_vector_type(8)))  float  v8f;
typedef unsigned short ushort_t;

union BF16x16 { v16bf v; unsigned int u[8]; };

__device__ __forceinline__ ushort_t f2bf(float f) {
    unsigned int x = __float_as_uint(f);
    unsigned int r = (x + 0x7FFFu + ((x >> 16) & 1u)) >> 16;
    return (ushort_t)r;
}
__device__ __forceinline__ unsigned int pack2(float a, float b) {
    return (unsigned int)f2bf(a) | ((unsigned int)f2bf(b) << 16);
}
__device__ __forceinline__ float wave_sum(float v) {
    for (int o = 16; o > 0; o >>= 1) v += __shfl_xor(v, o, 32);
    return v;
}
__device__ __forceinline__ float gelu_exact(float x) {
    return 0.5f * x * (1.0f + erff(x * 0.70710678118654752f));
}

// CDNA5 async global->LDS copy (ASYNCcnt-tracked), 16 bytes per lane.
__device__ __forceinline__ void async_copy_b128(ushort_t* ldsDst, const ushort_t* gSrc) {
    unsigned lds_off = (unsigned)(unsigned long long)ldsDst;       // low 32 bits = LDS offset
    unsigned long long ga = (unsigned long long)gSrc;
    asm volatile("global_load_async_to_lds_b128 %0, %1, off"
                 :: "v"(lds_off), "v"(ga) : "memory");
}
__device__ __forceinline__ void wait_async0() {
    asm volatile("s_wait_asynccnt 0x0" ::: "memory");
}

// ---------------------------------------------------------------------------
// GEMM, bf16 operands, f32 accumulate: C[M,N] = A[M,K] @ B[K,N] (+bias)
// 256 threads (8 waves, 2x4), tile 128x128, K step 32.
// Requires: M % 128 == 0, K % 32 == 0, lda % 8 == 0 (true for all call sites).
// ---------------------------------------------------------------------------
__global__ __launch_bounds__(256) void k_gemm_bias(
    const ushort_t* __restrict__ A, const ushort_t* __restrict__ Bm,
    const float* __restrict__ bias, float* __restrict__ C,
    int M, int N, int K, int lda, int ldb, int ldc)
{
    __shared__ __attribute__((aligned(16))) ushort_t sA[128 * 36]; // [m][k]
    __shared__ __attribute__((aligned(16))) ushort_t sB[128 * 36]; // [n][k] (transposed)

    const int tid  = threadIdx.x;
    const int lane = tid & 31;
    const int wave = tid >> 5;
    const int wm   = wave >> 2;          // 0..1
    const int wn   = wave & 3;           // 0..3
    const int lmod = lane & 15;
    const int lhalf = (lane >> 4) & 1;
    const int bm0 = blockIdx.y * 128;
    const int bn0 = blockIdx.x * 128;
    const bool vecB = ((ldb & 3) == 0) && (bn0 + 128 <= N);

    v8f acc[4][2];
    for (int s = 0; s < 4; ++s)
        for (int t = 0; t < 2; ++t)
            for (int r = 0; r < 8; ++r) acc[s][t][r] = 0.f;

    for (int k0 = 0; k0 < K; k0 += 32) {
        // ---- A tile: async copy 128x32 bf16 (512 x 16-byte chunks, 2/thread)
        for (int e = tid; e < 512; e += 256) {
            int m = e >> 2, q = e & 3;
            async_copy_b128(&sA[m * 36 + q * 8],
                            A + (size_t)(bm0 + m) * lda + k0 + q * 8);
        }
        // ---- B tile: vectorized read, transposed store into LDS
        if (vecB) {
            for (int e = tid; e < 1024; e += 256) {
                int kk = e >> 5, n4 = (e & 31) * 4;
                const uint2 v = *(const uint2*)(Bm + (size_t)(k0 + kk) * ldb + bn0 + n4);
                sB[(n4 + 0) * 36 + kk] = (ushort_t)(v.x);
                sB[(n4 + 1) * 36 + kk] = (ushort_t)(v.x >> 16);
                sB[(n4 + 2) * 36 + kk] = (ushort_t)(v.y);
                sB[(n4 + 3) * 36 + kk] = (ushort_t)(v.y >> 16);
            }
        } else {
            for (int e = tid; e < 4096; e += 256) {
                int kk = e >> 7, n = e & 127;
                int gn = bn0 + n;
                int gc = gn < N ? gn : (N - 1);
                ushort_t v = Bm[(size_t)(k0 + kk) * ldb + gc];
                if (gn >= N) v = 0;
                sB[n * 36 + kk] = v;
            }
        }
        wait_async0();
        __syncthreads();

        BF16x16 bfr[2];
        for (int t = 0; t < 2; ++t) {
            int n = wn * 32 + t * 16 + lmod;
            const unsigned int* p = (const unsigned int*)&sB[n * 36 + lhalf * 16];
            for (int j = 0; j < 8; ++j) bfr[t].u[j] = p[j];
        }
        for (int s = 0; s < 4; ++s) {
            BF16x16 afr;
            int m = wm * 64 + s * 16 + lmod;
            const ushort_t* pm = &sA[m * 36];
            for (int j = 0; j < 8; ++j) {
                int kOff = lhalf * 16 + 2 * j;
                afr.u[j] = *(const unsigned int*)&pm[kOff];
            }
            for (int t = 0; t < 2; ++t)
                acc[s][t] = __builtin_amdgcn_wmma_f32_16x16x32_bf16(
                    false, afr.v, false, bfr[t].v, (short)0, acc[s][t], false, false);
        }
        __syncthreads();
    }

    for (int s = 0; s < 4; ++s)
        for (int t = 0; t < 2; ++t) {
            int col = bn0 + wn * 32 + t * 16 + lmod;
            if (col >= N) continue;
            float bv = bias ? bias[col] : 0.f;
            for (int r = 0; r < 8; ++r) {
                int row = bm0 + wm * 64 + s * 16 + r + lhalf * 8;
                C[(size_t)row * ldc + col] = acc[s][t][r] + bv;
            }
        }
}

// ---------------------------------------------------------------------------
// Clustered attention: one block per (b, h, cluster). 64x64 attention via WMMA.
// ---------------------------------------------------------------------------
__global__ __launch_bounds__(256) void k_cluster_attn(
    const float* __restrict__ Q,    // raw q, [B*Tq, 512]
    const float* __restrict__ Kg,   // k-source (qn for self, raw k for cross)
    const float* __restrict__ V,    // raw v, [B*Tk, 512]
    const int*  __restrict__ qidx,  // [B*H*NCLUS, 64]
    const int*  __restrict__ kidx,
    float* __restrict__ num,        // [B*Tq*512]
    float* __restrict__ den,        // [B*H*Tq]
    int Tq, int Tk, int causal, int selfmask)
{
    __shared__ __attribute__((aligned(16))) ushort_t sQ [64 * 68];
    __shared__ __attribute__((aligned(16))) ushort_t sK [64 * 68];
    __shared__ __attribute__((aligned(16))) ushort_t sVt[64 * 68]; // [d][j]
    __shared__ __attribute__((aligned(16))) ushort_t sAt[64 * 68];
    __shared__ float sD[64 * 68];
    __shared__ int sQP[64], sKP[64];

    const int tid = threadIdx.x, lane = tid & 31, wave = tid >> 5;
    const int lmod = lane & 15, lhalf = (lane >> 4) & 1;
    const int bid = blockIdx.x;
    const int h = (bid >> 2) & 7, b = bid >> 5;

    if (tid < 64) sQP[tid] = qidx[bid * 64 + tid];
    else if (tid < 128) sKP[tid - 64] = kidx[bid * 64 + (tid - 64)];
    __syncthreads();

    for (int e4 = tid; e4 < 1024; e4 += 256) {
        int i = e4 >> 4, d4 = (e4 & 15) * 4;
        const float4 q4 = *(const float4*)(Q  + ((size_t)(b * Tq + sQP[i])) * 512 + h * 64 + d4);
        const float4 k4 = *(const float4*)(Kg + ((size_t)(b * Tk + sKP[i])) * 512 + h * 64 + d4);
        const float4 v4 = *(const float4*)(V  + ((size_t)(b * Tk + sKP[i])) * 512 + h * 64 + d4);
        *(uint2*)&sQ[i * 68 + d4] = make_uint2(pack2(q4.x, q4.y), pack2(q4.z, q4.w));
        *(uint2*)&sK[i * 68 + d4] = make_uint2(pack2(k4.x, k4.y), pack2(k4.z, k4.w));
        sVt[(d4 + 0) * 68 + i] = f2bf(v4.x);
        sVt[(d4 + 1) * 68 + i] = f2bf(v4.y);
        sVt[(d4 + 2) * 68 + i] = f2bf(v4.z);
        sVt[(d4 + 3) * 68 + i] = f2bf(v4.w);
    }
    __syncthreads();

    // dots = qg @ kg^T (scale 1/8), masks, -> sD
    for (int ti = 0; ti < 2; ++ti) {
        int tile = wave * 2 + ti, tm = tile >> 2, tn = tile & 3;
        v8f acc;
        for (int r = 0; r < 8; ++r) acc[r] = 0.f;
        for (int kc = 0; kc < 2; ++kc) {
            int k0 = kc * 32;
            BF16x16 afr, bfr;
            {
                int m = tm * 16 + lmod;
                const ushort_t* pm = &sQ[m * 68 + k0];
                for (int j = 0; j < 8; ++j) {
                    int kOff = lhalf * 16 + 2 * j;
                    afr.u[j] = *(const unsigned int*)&pm[kOff];
                }
            }
            {
                int n = tn * 16 + lmod;
                const unsigned int* p = (const unsigned int*)&sK[n * 68 + k0 + lhalf * 16];
                for (int j = 0; j < 8; ++j) bfr.u[j] = p[j];
            }
            acc = __builtin_amdgcn_wmma_f32_16x16x32_bf16(
                false, afr.v, false, bfr.v, (short)0, acc, false, false);
        }
        for (int r = 0; r < 8; ++r) {
            int i = tm * 16 + r + lhalf * 8, j = tn * 16 + lmod;
            float v = acc[r] * 0.125f;
            int qp = sQP[i], kp = sKP[j];
            if (causal && qp < kp) v = -1.0e9f;
            if (selfmask && qp == kp) v = -50000.0f;
            sD[i * 68 + j] = v;
        }
    }
    __syncthreads();

    // softmax rows -> bf16 sAt (4 threads per row)
    {
        int row = tid >> 2, part = tid & 3;
        float* rd = &sD[row * 68 + part * 16];
        float mx = -3.4e38f;
        for (int k = 0; k < 16; ++k) mx = fmaxf(mx, rd[k]);
        mx = fmaxf(mx, __shfl_xor(mx, 1, 32));
        mx = fmaxf(mx, __shfl_xor(mx, 2, 32));
        float ev[16], sum = 0.f;
        for (int k = 0; k < 16; ++k) { ev[k] = expf(rd[k] - mx); sum += ev[k]; }
        sum += __shfl_xor(sum, 1, 32);
        sum += __shfl_xor(sum, 2, 32);
        float inv = 1.f / sum;
        ushort_t* ra = &sAt[row * 68 + part * 16];
        for (int k = 0; k < 16; ++k) ra[k] = f2bf(ev[k] * inv);
    }
    __syncthreads();

    // o = attn @ vg, scatter-add
    for (int ti = 0; ti < 2; ++ti) {
        int tile = wave * 2 + ti, tm = tile >> 2, tn = tile & 3;
        v8f acc;
        for (int r = 0; r < 8; ++r) acc[r] = 0.f;
        for (int kc = 0; kc < 2; ++kc) {
            int k0 = kc * 32;
            BF16x16 afr, bfr;
            {
                int m = tm * 16 + lmod;
                const ushort_t* pm = &sAt[m * 68 + k0];
                for (int j = 0; j < 8; ++j) {
                    int kOff = lhalf * 16 + 2 * j;
                    afr.u[j] = *(const unsigned int*)&pm[kOff];
                }
            }
            {
                int n = tn * 16 + lmod; // n == output dim d
                const unsigned int* p = (const unsigned int*)&sVt[n * 68 + k0 + lhalf * 16];
                for (int j = 0; j < 8; ++j) bfr.u[j] = p[j];
            }
            acc = __builtin_amdgcn_wmma_f32_16x16x32_bf16(
                false, afr.v, false, bfr.v, (short)0, acc, false, false);
        }
        for (int r = 0; r < 8; ++r) {
            int i = tm * 16 + r + lhalf * 8, d = tn * 16 + lmod;
            atomicAdd(&num[((size_t)(b * Tq + sQP[i])) * 512 + h * 64 + d], acc[r]);
        }
    }
    if (tid < 64) atomicAdd(&den[(size_t)(b * HEADS_ + h) * Tq + sQP[tid]], 1.0f);
}

// ---------------------------------------------------------------------------
// LayerNorm, one wave per row (rows = 16384). f32-out (residual) variant and
// bf16-out (GEMM feed) variant with optional fused GELU.
// ---------------------------------------------------------------------------
__global__ __launch_bounds__(256) void k_ln_f32(
    const float* __restrict__ X, const float* __restrict__ g,
    const float* __restrict__ bta, float* __restrict__ Y, int cols)
{
    int row = blockIdx.x * 8 + (threadIdx.x >> 5);
    int lane = threadIdx.x & 31;
    const float* x = X + (size_t)row * cols;
    float s = 0.f, s2 = 0.f;
    for (int c = lane; c < cols; c += 32) { float v = x[c]; s += v; s2 += v * v; }
    s = wave_sum(s); s2 = wave_sum(s2);
    float mean = s / cols;
    float var = fmaxf(s2 / cols - mean * mean, 0.f);
    float inv = rsqrtf(var + 1e-5f);
    float* y = Y + (size_t)row * cols;
    for (int c = lane; c < cols; c += 32)
        y[c] = (x[c] - mean) * inv * g[c] + bta[c];
}

__global__ __launch_bounds__(256) void k_ln_bf16(
    const float* __restrict__ X, const float* __restrict__ g,
    const float* __restrict__ bta, ushort_t* __restrict__ Y, int cols, int dogelu)
{
    int row = blockIdx.x * 8 + (threadIdx.x >> 5);
    int lane = threadIdx.x & 31;
    const float* x = X + (size_t)row * cols;
    float s = 0.f, s2 = 0.f;
    for (int c = lane; c < cols; c += 32) { float v = x[c]; s += v; s2 += v * v; }
    s = wave_sum(s); s2 = wave_sum(s2);
    float mean = s / cols;
    float var = fmaxf(s2 / cols - mean * mean, 0.f);
    float inv = rsqrtf(var + 1e-5f);
    ushort_t* y = Y + (size_t)row * cols;
    for (int c = lane; c < cols; c += 32) {
        float v = (x[c] - mean) * inv * g[c] + bta[c];
        if (dogelu) v = gelu_exact(v);
        y[c] = f2bf(v);
    }
}

// ---------------------------------------------------------------------------
// Routing: per (b,h,t) wave: l2-normalize head, cluster scores, commit aux
// ---------------------------------------------------------------------------
__global__ __launch_bounds__(256) void k_route(
    const float* __restrict__ X, const float* __restrict__ means,
    float* __restrict__ Xn, float* __restrict__ scores,
    float* __restrict__ aux, int Tt, float auxScale)
{
    int gid = blockIdx.x * 8 + (threadIdx.x >> 5);
    int lane = threadIdx.x & 31;
    if (gid >= B_ * HEADS_ * Tt) return;
    int t = gid % Tt, bh = gid / Tt;
    int h = bh % HEADS_, b = bh / HEADS_;

    const float* x = X + ((size_t)(b * Tt + t)) * 512 + h * 64;
    float v0 = x[lane], v1 = x[lane + 32];
    float nrm = sqrtf(wave_sum(v0 * v0 + v1 * v1));
    float inv = 1.0f / fmaxf(nrm, 1e-12f);
    float q0 = v0 * inv, q1 = v1 * inv;
    float* xn = Xn + ((size_t)(b * Tt + t)) * 512 + h * 64;
    xn[lane] = q0; xn[lane + 32] = q1;

    float best = -3.4e38f; int bc = 0;
    for (int c = 0; c < NCLUS_; ++c) {
        const float* m = means + (size_t)(h * NCLUS_ + c) * 64;
        float sc = wave_sum(q0 * m[lane] + q1 * m[lane + 32]);
        if (lane == 0) scores[((size_t)(b * HEADS_ + h) * NCLUS_ + c) * Tt + t] = sc;
        if (sc > best) { best = sc; bc = c; }
    }
    const float* m = means + (size_t)(h * NCLUS_ + bc) * 64;
    float d0 = q0 - m[lane], d1 = q1 - m[lane + 32];
    float se = wave_sum(d0 * d0 + d1 * d1);
    if (lane == 0) atomicAdd(aux, se * auxScale);
}

// ---------------------------------------------------------------------------
// Top-64 of 256 scores, one wave per (b,h,c)
// ---------------------------------------------------------------------------
__global__ __launch_bounds__(32) void k_topk(
    const float* __restrict__ scores, int* __restrict__ idx, int Tt)
{
    int lane = threadIdx.x;
    const float* row = scores + (size_t)blockIdx.x * Tt;
    float v[8];
    for (int j = 0; j < 8; ++j) {
        int t = lane + j * 32;
        v[j] = (t < Tt) ? row[t] : -3.4e38f;
    }
    for (int s = 0; s < WSZ_; ++s) {
        float bv = -3.4e38f; int bi = 0x7fffffff;
        for (int j = 0; j < 8; ++j) {
            int t = lane + j * 32;
            if (v[j] > bv || (v[j] == bv && t < bi)) { bv = v[j]; bi = t; }
        }
        for (int o = 16; o > 0; o >>= 1) {
            float ov = __shfl_xor(bv, o, 32);
            int oi = __shfl_xor(bi, o, 32);
            if (ov > bv || (ov == bv && oi < bi)) { bv = ov; bi = oi; }
        }
        if (lane == 0) idx[(size_t)blockIdx.x * WSZ_ + s] = bi;
        for (int j = 0; j < 8; ++j)
            if (lane + j * 32 == bi) v[j] = -3.4e38f;
    }
}

// ---------------------------------------------------------------------------
// Elementwise helpers
// ---------------------------------------------------------------------------
__global__ void k_cvt_bf16(const float* __restrict__ s, ushort_t* __restrict__ d, size_t n)
{
    for (size_t e = blockIdx.x * 256ull + threadIdx.x; e < n; e += gridDim.x * 256ull)
        d[e] = f2bf(s[e]);
}

__global__ void k_embed(const float* __restrict__ saml, const float* __restrict__ emb,
                        ushort_t* __restrict__ out, size_t n)
{
    for (size_t e = blockIdx.x * 256ull + threadIdx.x; e < n; e += gridDim.x * 256ull) {
        size_t bt = e >> 9; int c = (int)(e & 511);
        float v;
        if (c < EMB_DIM_) {
            int id = (int)saml[bt * 13];
            v = emb[(size_t)id * EMB_DIM_ + c];
        } else {
            v = saml[bt * 13 + 1 + (c - EMB_DIM_)];
        }
        out[e] = f2bf(v);
    }
}

__global__ void k_add_ip(float* __restrict__ d, const float* __restrict__ s, size_t n)
{
    for (size_t e = blockIdx.x * 256ull + threadIdx.x; e < n; e += gridDim.x * 256ull)
        d[e] += s[e];
}

__global__ void k_avg(const float* __restrict__ a, const float* __restrict__ b,
                      float* __restrict__ o, size_t n)
{
    for (size_t e = blockIdx.x * 256ull + threadIdx.x; e < n; e += gridDim.x * 256ull)
        o[e] = 0.5f * (a[e] + b[e]);
}

__global__ void k_relu_ip(float* __restrict__ x, size_t n)
{
    for (size_t e = blockIdx.x * 256ull + threadIdx.x; e < n; e += gridDim.x * 256ull)
        x[e] = fmaxf(x[e], 0.f);
}

// u[16384,4096] f32 -> Gh[16384,2048] bf16:  G = a * gelu(g)
__global__ void k_gate(const float* __restrict__ u, ushort_t* __restrict__ G, size_t n)
{
    for (size_t e = blockIdx.x * 256ull + threadIdx.x; e < n; e += gridDim.x * 256ull) {
        size_t i = e >> 11; int j = (int)(e & 2047);
        float a = u[i * 4096 + j];
        float g = u[i * 4096 + 2048 + j];
        G[e] = f2bf(a * gelu_exact(g));
    }
}

__global__ void k_attn_finalize(const float* __restrict__ num, const float* __restrict__ den,
                                ushort_t* __restrict__ out, size_t n)
{
    for (size_t e = blockIdx.x * 256ull + threadIdx.x; e < n; e += gridDim.x * 256ull) {
        size_t bt = e >> 9; int col = (int)(e & 511);
        int h = col >> 6;
        int t = (int)(bt & (T_ - 1));
        int b = (int)(bt >> 8);
        float d = den[((size_t)(b * HEADS_ + h)) * T_ + t];
        out[e] = f2bf(num[e] / (d + 1e-5f));
    }
}

// ---------------------------------------------------------------------------
// Host orchestration
// ---------------------------------------------------------------------------
extern "C" void kernel_launch(void* const* d_in, const int* in_sizes, int n_in,
                              void* d_out, int out_size, void* d_ws, size_t ws_size,
                              hipStream_t stream)
{
    (void)in_sizes; (void)n_in; (void)out_size; (void)ws_size;

    // ----- parameter walk (setup_inputs insertion order, recursively) -----
    int cur = 0;
    auto F = [&]() -> const float* { return (const float*)d_in[cur++]; };
    const float* saml    = F();   // [B,T,13]
    const float* context = F();   // [B,CTX,512]
    const float* emb     = F();   // [256,500]
    const float* pre_w = F(); const float* pre_b = F();
    const float* pre_g = F(); const float* pre_be = F();
    const float* post_g = F(); const float* post_be = F();

    struct AttnP { const float *wq,*wv,*wo,*bo,*means,*wk; };
    struct FFP   { const float *w1,*b1,*w2,*b2; };
    struct LayerP { const float *n1g,*n1b,*n2g,*n2b,*n3g,*n3b,*n4g,*n4b; AttnP attn, cross; FFP ff1, ff2; };
    LayerP L[DEPTH_];
    for (int i = 0; i < DEPTH_; ++i) {
        L[i].n1g = F(); L[i].n1b = F();
        L[i].attn.wq = F(); L[i].attn.wv = F(); L[i].attn.wo = F();
        L[i].attn.bo = F(); L[i].attn.means = F(); L[i].attn.wk = nullptr;
        L[i].n2g = F(); L[i].n2b = F();
        L[i].ff1.w1 = F(); L[i].ff1.b1 = F(); L[i].ff1.w2 = F(); L[i].ff1.b2 = F();
        L[i].n3g = F(); L[i].n3b = F();
        L[i].cross.wq = F(); L[i].cross.wv = F(); L[i].cross.wo = F();
        L[i].cross.bo = F(); L[i].cross.means = F(); L[i].cross.wk = F();
        L[i].n4g = F(); L[i].n4b = F();
        L[i].ff2.w1 = F(); L[i].ff2.b1 = F(); L[i].ff2.w2 = F(); L[i].ff2.b2 = F();
    }
    struct HeadP { const float *w1,*b1,*g1,*be1,*w2,*b2,*g2,*be2,*w3,*b3; };
    HeadP hcls, hcol, hpos;
    auto loadHead = [&](HeadP& hp) {
        hp.w1 = F(); hp.b1 = F(); hp.g1 = F(); hp.be1 = F();
        hp.w2 = F(); hp.b2 = F(); hp.g2 = F(); hp.be2 = F();
        hp.w3 = F(); hp.b3 = F();
    };
    loadHead(hcls); loadHead(hcol); loadHead(hpos);

    // ----- workspace arena (byte cursor, 16B aligned carves) -----
    char* base = (char*)d_ws;
    size_t off = 0;
    auto allocB = [&](size_t bytes) { void* p = base + off; off = (off + bytes + 15) & ~(size_t)15; return p; };
    const size_t ACT = (size_t)MT_ * DIM_;       // 16384 * 512

    float* x1   = (float*)allocB(ACT * 4);
    float* x2   = (float*)allocB(ACT * 4);
    float* qb   = (float*)allocB(ACT * 4);
    float* vb   = (float*)allocB(ACT * 4);
    float* kb   = (float*)allocB(ACT * 4);
    float* qn   = (float*)allocB(ACT * 4);
    float* knb  = (float*)allocB(ACT * 4);
    float* numb = (float*)allocB(ACT * 4);
    float* U    = (float*)allocB((size_t)MT_ * 4096 * 4);   // FF mid / head mid (f32)
    float* den  = (float*)allocB((size_t)B_ * HEADS_ * T_ * 4);
    float* scq  = (float*)allocB((size_t)B_ * HEADS_ * NCLUS_ * T_ * 4);
    float* sck  = (float*)allocB((size_t)B_ * HEADS_ * NCLUS_ * T_ * 4);
    int* qidx   = (int*)allocB((size_t)B_ * HEADS_ * NCLUS_ * WSZ_ * 4);
    int* kidx   = (int*)allocB((size_t)B_ * HEADS_ * NCLUS_ * WSZ_ * 4);
    ushort_t* t0h   = (ushort_t*)allocB(ACT * 2);           // LN/GEMM-feed (bf16)
    ushort_t* tmrgh = (ushort_t*)allocB(ACT * 2);           // merged attn / embed (bf16)
    ushort_t* ctxh  = (ushort_t*)allocB(ACT * 2);           // context (bf16)
    ushort_t* Gh    = (ushort_t*)allocB((size_t)MT_ * 2048 * 2); // gate out / head mid (bf16)
    ushort_t* hBh   = Gh;                                   // heads run after layers
    float* hA = U;

    float* outF = (float*)d_out;
    float* auxp = outF + (size_t)MT_ * 512;      // 500+4+8 == 512 per token, aux last

    // bf16 weight arena
    auto cvt = [&](const float* src, size_t n) -> const ushort_t* {
        ushort_t* d = (ushort_t*)allocB(n * 2);
        k_cvt_bf16<<<dim3(512), dim3(256), 0, stream>>>(src, d, n);
        return d;
    };
    const ushort_t* pre_wh = cvt(pre_w, (size_t)DIM_ * DIM_);
    struct AttnH { const ushort_t *wq,*wv,*wo,*wk; };
    struct FFH   { const ushort_t *w1,*w2; };
    struct LayerH { AttnH attn, cross; FFH ff1, ff2; };
    LayerH LH[DEPTH_];
    for (int i = 0; i < DEPTH_; ++i) {
        LH[i].attn.wq = cvt(L[i].attn.wq, (size_t)DIM_ * DIM_);
        LH[i].attn.wv = cvt(L[i].attn.wv, (size_t)DIM_ * DIM_);
        LH[i].attn.wo = cvt(L[i].attn.wo, (size_t)DIM_ * DIM_);
        LH[i].attn.wk = nullptr;
        LH[i].ff1.w1  = cvt(L[i].ff1.w1, (size_t)DIM_ * 4096);
        LH[i].ff1.w2  = cvt(L[i].ff1.w2, (size_t)2048 * DIM_);
        LH[i].cross.wq = cvt(L[i].cross.wq, (size_t)DIM_ * DIM_);
        LH[i].cross.wk = cvt(L[i].cross.wk, (size_t)DIM_ * DIM_);
        LH[i].cross.wv = cvt(L[i].cross.wv, (size_t)DIM_ * DIM_);
        LH[i].cross.wo = cvt(L[i].cross.wo, (size_t)DIM_ * DIM_);
        LH[i].ff2.w1  = cvt(L[i].ff2.w1, (size_t)DIM_ * 4096);
        LH[i].ff2.w2  = cvt(L[i].ff2.w2, (size_t)2048 * DIM_);
    }
    struct HeadH { const ushort_t *w1,*w2,*w3; };
    HeadH hclsH = { cvt(hcls.w1, (size_t)DIM_*1024), cvt(hcls.w2, (size_t)1024*1024), cvt(hcls.w3, (size_t)1024*500) };
    HeadH hcolH = { cvt(hcol.w1, (size_t)DIM_*1024), cvt(hcol.w2, (size_t)1024*1024), cvt(hcol.w3, (size_t)1024*4) };
    HeadH hposH = { cvt(hpos.w1, (size_t)DIM_*1024), cvt(hpos.w2, (size_t)1024*1024), cvt(hpos.w3, (size_t)1024*8) };
    k_cvt_bf16<<<dim3(512), dim3(256), 0, stream>>>(context, ctxh, ACT);

    // ----- launch helpers -----
    auto gemm = [&](const ushort_t* A, const ushort_t* Bm, const float* bias, float* C,
                    int M, int N, int K, int lda, int ldb, int ldc) {
        dim3 g((N + 127) / 128, (M + 127) / 128);
        k_gemm_bias<<<g, dim3(256), 0, stream>>>(A, Bm, bias, C, M, N, K, lda, ldb, ldc);
    };
    auto lnf = [&](const float* X, const float* g, const float* b, float* Y, int cols) {
        k_ln_f32<<<dim3(MT_ / 8), dim3(256), 0, stream>>>(X, g, b, Y, cols);
    };
    auto lnh = [&](const float* X, const float* g, const float* b, ushort_t* Y, int cols, int dogelu) {
        k_ln_bf16<<<dim3(MT_ / 8), dim3(256), 0, stream>>>(X, g, b, Y, cols, dogelu);
    };
    auto addip = [&](float* d, const float* s, size_t n) {
        k_add_ip<<<dim3(4096), dim3(256), 0, stream>>>(d, s, n);
    };
    const float selfScale  = COMMIT_ / (float)((size_t)B_ * HEADS_ * T_ * DH_);  // both halves identical
    const float crossScale = COMMIT_ / (float)((size_t)B_ * HEADS_ * (T_ + CTX_) * DH_);

    auto route = [&](const float* X, const float* means, float* Xn, float* sc, float scale) {
        k_route<<<dim3((B_ * HEADS_ * T_) / 8), dim3(256), 0, stream>>>(X, means, Xn, sc, auxp, T_, scale);
    };
    auto topk = [&](const float* sc, int* id) {
        k_topk<<<dim3(B_ * HEADS_ * NCLUS_), dim3(32), 0, stream>>>(sc, id, T_);
    };
    auto attn = [&](const float* Q, const float* Kg, const float* V,
                    const int* qi, const int* ki, int causal, int selfm) {
        hipMemsetAsync(numb, 0, ACT * sizeof(float), stream);
        hipMemsetAsync(den, 0, (size_t)B_ * HEADS_ * T_ * sizeof(float), stream);
        k_cluster_attn<<<dim3(B_ * HEADS_ * NCLUS_), dim3(256), 0, stream>>>(
            Q, Kg, V, qi, ki, numb, den, T_, T_, causal, selfm);
        k_attn_finalize<<<dim3(4096), dim3(256), 0, stream>>>(numb, den, tmrgh, ACT);
    };
    auto ff = [&](float* xin_src, const float* ng, const float* nb, const FFP& p, const FFH& ph, float* xdst) {
        lnh(xin_src, ng, nb, t0h, DIM_, 0);
        gemm(t0h, ph.w1, p.b1, U, MT_, 4096, DIM_, DIM_, 4096, 4096);
        k_gate<<<dim3(4096), dim3(256), 0, stream>>>(U, Gh, (size_t)MT_ * 2048);
        gemm(Gh, ph.w2, p.b2, numb, MT_, DIM_, 2048, 2048, DIM_, DIM_);
        addip(xdst, numb, ACT);
    };

    // ----- forward pass -----
    hipMemsetAsync(auxp, 0, sizeof(float), stream);

    // embed + pre-projection + pre-LN
    k_embed<<<dim3(4096), dim3(256), 0, stream>>>(saml, emb, tmrgh, ACT);
    gemm(tmrgh, pre_wh, pre_b, qb, MT_, DIM_, DIM_, DIM_, DIM_, DIM_);
    lnf(qb, pre_g, pre_be, x1, DIM_);
    hipMemcpyAsync(x2, x1, ACT * sizeof(float), hipMemcpyDeviceToDevice, stream);

    for (int i = 0; i < DEPTH_; ++i) {
        // --- self attention on ln(x2) ---
        lnh(x2, L[i].n1g, L[i].n1b, t0h, DIM_, 0);
        gemm(t0h, LH[i].attn.wq, nullptr, qb, MT_, DIM_, DIM_, DIM_, DIM_, DIM_);
        gemm(t0h, LH[i].attn.wv, nullptr, vb, MT_, DIM_, DIM_, DIM_, DIM_, DIM_);
        route(qb, L[i].attn.means, qn, scq, selfScale);
        topk(scq, qidx);
        attn(qb, qn, vb, qidx, qidx, /*causal*/1, /*selfmask*/1);
        gemm(tmrgh, LH[i].attn.wo, L[i].attn.bo, numb, MT_, DIM_, DIM_, DIM_, DIM_, DIM_);
        addip(x1, numb, ACT);

        // --- ff1 on ln(x1) -> x2 ---
        ff(x1, L[i].n2g, L[i].n2b, L[i].ff1, LH[i].ff1, x2);

        // --- cross attention on ln(x2), context ---
        lnh(x2, L[i].n3g, L[i].n3b, t0h, DIM_, 0);
        gemm(t0h, LH[i].cross.wq, nullptr, qb, MT_, DIM_, DIM_, DIM_, DIM_, DIM_);
        gemm(ctxh, LH[i].cross.wk, nullptr, kb, MT_, DIM_, DIM_, DIM_, DIM_, DIM_);
        gemm(ctxh, LH[i].cross.wv, nullptr, vb, MT_, DIM_, DIM_, DIM_, DIM_, DIM_);
        route(qb, L[i].cross.means, qn, scq, crossScale);
        route(kb, L[i].cross.means, knb, sck, crossScale);
        topk(scq, qidx);
        topk(sck, kidx);
        attn(qb, kb, vb, qidx, kidx, /*causal*/0, /*selfmask*/0);
        gemm(tmrgh, LH[i].cross.wo, L[i].cross.bo, numb, MT_, DIM_, DIM_, DIM_, DIM_, DIM_);
        addip(x1, numb, ACT);

        // --- ff2 on ln(x1) -> x2 ---
        ff(x1, L[i].n4g, L[i].n4b, L[i].ff2, LH[i].ff2, x2);
    }

    // post: ln(0.5*(x1+x2)) -> bf16 head input
    k_avg<<<dim3(4096), dim3(256), 0, stream>>>(x1, x2, qb, ACT);
    lnh(qb, post_g, post_be, t0h, DIM_, 0);

    // heads
    auto runHead = [&](const HeadP& hp, const HeadH& hh, float* op, int N3, bool dorelu) {
        gemm(t0h, hh.w1, hp.b1, hA, MT_, 1024, DIM_, DIM_, 1024, 1024);
        lnh(hA, hp.g1, hp.be1, hBh, 1024, 1);
        gemm(hBh, hh.w2, hp.b2, hA, MT_, 1024, 1024, 1024, 1024, 1024);
        lnh(hA, hp.g2, hp.be2, hBh, 1024, 1);
        gemm(hBh, hh.w3, hp.b3, op, MT_, N3, 1024, 1024, N3, N3);
        if (dorelu) k_relu_ip<<<dim3(4096), dim3(256), 0, stream>>>(op, (size_t)MT_ * N3);
    };
    runHead(hcls, hclsH, outF, 500, false);
    runHead(hcol, hcolH, outF + (size_t)MT_ * 500, 4, true);
    runHead(hpos, hposH, outF + (size_t)MT_ * 504, 8, true);
}
